// MaskedLPropDecoderLayer_56951266345700
// MI455X (gfx1250) — matrix operand order repaired and checked by reading, not compiled, not run on hardware
//
#include <hip/hip_runtime.h>
#include <hip/hip_bf16.h>

// ---------------- problem constants ----------------
#define NF    8           // frames
#define NB    2           // batch
#define CDIM  384         // channels
#define HW    576         // 24*24
#define KD    128         // key dim
#define S_TOT 4608        // NF*HW

typedef __attribute__((ext_vector_type(16))) _Float16 v16h;
typedef __attribute__((ext_vector_type(8)))  _Float16 v8h;
typedef __attribute__((ext_vector_type(8)))  float    v8f;

__device__ __forceinline__ v16h ld_op32(const _Float16* p0, const _Float16* p1) {
    v8h a0 = *(const v8h*)p0;
    v8h a1 = *(const v8h*)p1;
    v16h r;
#pragma unroll
    for (int i = 0; i < 8; ++i) { r[i] = a0[i]; r[8 + i] = a1[i]; }
    return r;
}

// ============================================================
// Cast input to f16 row-major XA[b][s][c].
// layoutBSC: 0 -> in is [F][B][C][HW] ; 1 -> in is [B][S][C]
// ============================================================
__global__ void xcast_kernel(const float* __restrict__ in, int layoutBSC,
                             _Float16* __restrict__ XA)
{
    const int total = NB * S_TOT * CDIM;
    int idx = blockIdx.x * blockDim.x + threadIdx.x;
    if (idx >= total) return;
    if (layoutBSC) { XA[idx] = (_Float16)in[idx]; return; }
    const int c = idx % CDIM;
    int t = idx / CDIM;
    const int s = t % S_TOT;
    const int b = t / S_TOT;
    const int f = s / HW, hw = s - f * HW;
    XA[idx] = (_Float16)in[(((size_t)(f * NB + b)) * CDIM + c) * HW + hw];
}

// ============================================================
// Weight prep: WB[cc][kd][j] = Wk[kd][cc*16+j]  (B-operand layout, f16)
// cc = 0..23 (C/16), kd = 0..127
// ============================================================
__global__ void wprep_kernel(const float* __restrict__ Wk,
                             _Float16* __restrict__ WB)
{
    const int total = (CDIM / 16) * KD * 16;
    int idx = blockIdx.x * blockDim.x + threadIdx.x;
    if (idx >= total) return;
    const int j = idx & 15;
    int t = idx >> 4;
    const int kd = t % KD;
    const int cc = t / KD;
    WB[idx] = (_Float16)Wk[(size_t)kd * CDIM + cc * 16 + j];
}

// ============================================================
// WMMA projection + L2 norm: out = l2norm(X @ Wk^T + b)
// One wave handles 16 rows x 128 kd: 8 accumulator tiles, K=384 (12 chunks).
// 4 waves / block. Writes QA [B][S][KD] and/or KB [B][8][S][16].
// ============================================================
__global__ __launch_bounds__(128) void projw_kernel(
    const _Float16* __restrict__ XA,   // [B][S][C] f16
    const _Float16* __restrict__ WB,   // [24][128][16] f16
    const float*    __restrict__ bk,   // [KD]
    _Float16* __restrict__ QA,         // [B][S][KD]   (nullable)
    _Float16* __restrict__ KBo)        // [B][8][S][16] (nullable)
{
    const int wave   = threadIdx.x >> 5;
    const int lane   = threadIdx.x & 31;
    const int lane15 = lane & 15;
    const int hi     = lane >> 4;
    const int qt     = blockIdx.x * 4 + wave;
    const int b      = blockIdx.y;
    const int q0     = qt * 16;

    v8f acc[8];
#pragma unroll
    for (int t = 0; t < 8; ++t) acc[t] = v8f{};

    const _Float16* xrow = XA + ((size_t)(b * S_TOT + q0 + lane15)) * CDIM;
    for (int kc = 0; kc < 12; ++kc) {
        v16h a = ld_op32(xrow + kc * 32 + hi * 16, xrow + kc * 32 + hi * 16 + 8);
        const _Float16* wbase = WB + (((size_t)(kc * 2 + hi)) * KD + lane15) * 16;
#pragma unroll
        for (int t = 0; t < 8; ++t) {
            const _Float16* wp = wbase + (size_t)t * 16 * 16;
            v16h bb = ld_op32(wp, wp + 8);
            acc[t] = __builtin_amdgcn_wmma_f32_16x16x32_f16(false, a, false, bb,
                                                            (short)0, acc[t], false, false);
        }
    }

    // bias (column = t*16 + lane15 in D layout) + per-row sum of squares
    float ss[8];
#pragma unroll
    for (int r = 0; r < 8; ++r) ss[r] = 0.f;
#pragma unroll
    for (int t = 0; t < 8; ++t) {
        const float bias = bk[t * 16 + lane15];
#pragma unroll
        for (int r = 0; r < 8; ++r) {
            acc[t][r] += bias;
            ss[r] = fmaf(acc[t][r], acc[t][r], ss[r]);
        }
    }
    // butterfly over the 16 lanes that share each row (masks 1..8 stay in-half)
#pragma unroll
    for (int m = 8; m >= 1; m >>= 1) {
#pragma unroll
        for (int r = 0; r < 8; ++r) ss[r] += __shfl_xor(ss[r], m, 32);
    }
    float inv[8];
#pragma unroll
    for (int r = 0; r < 8; ++r) inv[r] = 1.0f / fmaxf(sqrtf(ss[r]), 1e-12f);

#pragma unroll
    for (int t = 0; t < 8; ++t) {
        const int kd = t * 16 + lane15;
#pragma unroll
        for (int r = 0; r < 8; ++r) {
            const int s = q0 + r + 8 * hi;
            const _Float16 h = (_Float16)(acc[t][r] * inv[r]);
            if (QA)  QA[((size_t)(b * S_TOT + s)) * KD + kd] = h;
            if (KBo) KBo[(((size_t)(b * 8 + t)) * S_TOT + s) * 16 + lane15] = h;
        }
    }
}

// ============================================================
// V prep: VB[b][sc][c][j] = V[s=sc*16+j][b][c]  (f16); input [F][B][C][HW]
// ============================================================
__global__ void vprep_kernel(const float* __restrict__ in,
                             _Float16* __restrict__ VBo)
{
    const int total = NB * S_TOT * CDIM;
    int idx = blockIdx.x * blockDim.x + threadIdx.x;
    if (idx >= total) return;
    const int j = idx & 15;
    int t = idx >> 4;
    const int c = t % CDIM;  t /= CDIM;
    const int sc = t % (S_TOT / 16);
    const int b  = t / (S_TOT / 16);
    const int s  = sc * 16 + j;
    const int f  = s / HW;
    const int hw = s - f * HW;
    VBo[idx] = (_Float16)in[(((size_t)(f * NB + b)) * CDIM + c) * HW + hw];
}

// ============================================================
// Attention with block-structured frame mask (skip same-frame key blocks).
// 8 waves / block, wave w owns channels [w*48, w*48+48).
// Scores computed once per block by a rotating wave-pair, shared via a
// double-buffered A-layout LDS patch; ONE barrier per key block.
// Denominator: P x ones on wave 7, broadcast once at the end.
// ============================================================
__global__ __launch_bounds__(256) void attn_kernel(
    const _Float16* __restrict__ QA,   // [B][S][KD]
    const _Float16* __restrict__ KB,   // [B][8][S][16]
    const _Float16* __restrict__ VB,   // [B][S/16][C][16]
    const float*    __restrict__ res5d,// [F][B][C][HW] or nullptr
    float*          __restrict__ out)  // [B][S][C]
{
    __shared__ _Float16 pbuf[2][16 * 40];  // A-layout P patches (stride 40 halves)
    __shared__ float    dshare[16];

    const int tid    = threadIdx.x;
    const int wave   = tid >> 5;
    const int lane   = tid & 31;
    const int lane15 = lane & 15;
    const int hi     = lane >> 4;
    const int qt     = blockIdx.x;
    const int b      = blockIdx.y;
    const int q0     = qt * 16;
    const int fq     = q0 / HW;
    const int cbase  = wave * 48;

    // Q as A-operand chunks (KD=128 -> 4 chunks of k=32), kept resident
    v16h q[4];
    {
        const _Float16* qrow = QA + ((size_t)(b * S_TOT + q0 + lane15)) * KD;
#pragma unroll
        for (int kc = 0; kc < 4; ++kc)
            q[kc] = ld_op32(qrow + kc * 32 + hi * 16, qrow + kc * 32 + hi * 16 + 8);
    }

    v16h ones;
#pragma unroll
    for (int i = 0; i < 16; ++i) ones[i] = (_Float16)1.0f;

    v8f acc[3]; acc[0] = v8f{}; acc[1] = v8f{}; acc[2] = v8f{};
    v8f dacc = v8f{};

    const _Float16* kbB = KB + ((size_t)b * 8) * S_TOT * 16;
    const _Float16* vbB = VB + ((size_t)b * (S_TOT / 16)) * CDIM * 16;

    int it = 0;
    for (int kb = 0; kb < S_TOT / 32; ++kb) {
        const int key0 = kb * 32;
        if (key0 / HW == fq) continue;     // frame-masked block -> p == 0

        const int p = it & 1;

        // ---- rotating writer pair computes the 16x32 score block ----
        if ((wave >> 1) == (it & 3)) {
            const int tile = wave & 1;                     // which 16 keys
            const int kcol = key0 + tile * 16 + lane15;
            v16h bb[4];
#pragma unroll
            for (int kc = 0; kc < 4; ++kc) {
                const _Float16* kp = kbB + (((size_t)(2 * kc + hi)) * S_TOT + kcol) * 16;
                bb[kc] = ld_op32(kp, kp + 8);
            }
            v8f sc = v8f{};
#pragma unroll
            for (int kc = 0; kc < 4; ++kc)
                sc = __builtin_amdgcn_wmma_f32_16x16x32_f16(false, q[kc], false, bb[kc],
                                                            (short)0, sc, false, false);
            _Float16* pd = &pbuf[p][0];
#pragma unroll
            for (int r = 0; r < 8; ++r)
                pd[(r + 8 * hi) * 40 + tile * 16 + lane15] =
                    (_Float16)__expf(sc[r] * 30.0f);
        }

        // ---- hoist V-tile loads (independent of P) across the barrier ----
        const _Float16* vbase = vbB + ((size_t)(kb * 2 + hi)) * CDIM * 16;
        v16h vb[3];
#pragma unroll
        for (int t = 0; t < 3; ++t) {
            const _Float16* vp = vbase + (size_t)(cbase + t * 16 + lane15) * 16;
            vb[t] = ld_op32(vp, vp + 8);
        }
        __builtin_prefetch(kbB + ((size_t)hi * S_TOT + key0 + 32 + lane15) * 16, 0, 0);

        __syncthreads();

        // ---- all waves consume the shared P block ----
        v16h pa = ld_op32(&pbuf[p][lane15 * 40 + hi * 16],
                          &pbuf[p][lane15 * 40 + hi * 16 + 8]);
#pragma unroll
        for (int t = 0; t < 3; ++t)
            acc[t] = __builtin_amdgcn_wmma_f32_16x16x32_f16(false, pa, false, vb[t],
                                                            (short)0, acc[t], false, false);
        if (wave == 7)
            dacc = __builtin_amdgcn_wmma_f32_16x16x32_f16(false, pa, false, ones,
                                                          (short)0, dacc, false, false);
        ++it;
    }

    // ---- broadcast denominator row-sums ----
    if (wave == 7 && lane15 == 0) {
#pragma unroll
        for (int r = 0; r < 8; ++r) dshare[8 * hi + r] = dacc[r];
    }
    __syncthreads();
    float dn[8];
#pragma unroll
    for (int r = 0; r < 8; ++r) dn[r] = dshare[8 * hi + r];

    // ---- divide, add residual, store [B][S][C] ----
#pragma unroll
    for (int t = 0; t < 3; ++t) {
        const int ch = cbase + t * 16 + lane15;
#pragma unroll
        for (int r = 0; r < 8; ++r) {
            const int s = q0 + r + 8 * hi;
            float val = acc[t][r] / dn[r];
            if (res5d) {
                const int f = s / HW, hw = s - f * HW;
                val += res5d[(((size_t)(f * NB + b)) * CDIM + ch) * HW + hw];
            }
            out[((size_t)(b * S_TOT + s)) * CDIM + ch] = val;
        }
    }
}

// ============================================================
// InstanceNorm2d (affine=False, biased var, eps=1e-5) per (f,b,c)
// in: [B][S][C];  out5d: 0 -> [B][S][C], 1 -> [F][B][C][HW]
// ============================================================
__global__ __launch_bounds__(64) void inorm_kernel(
    const float* __restrict__ in, float* __restrict__ out, int out5d)
{
    const int gid = blockIdx.x;
    const int c = gid % CDIM;
    const int rest = gid / CDIM;
    const int b = rest % NB;
    const int f = rest / NB;

    __shared__ float ssum[64], ssq[64];
    float sum = 0.f, sq = 0.f;
    for (int hw = threadIdx.x; hw < HW; hw += 64) {
        float v = in[((size_t)(b * S_TOT + f * HW + hw)) * CDIM + c];
        sum += v; sq += v * v;
    }
    ssum[threadIdx.x] = sum; ssq[threadIdx.x] = sq;
    __syncthreads();
    for (int off = 32; off > 0; off >>= 1) {
        if (threadIdx.x < off) {
            ssum[threadIdx.x] += ssum[threadIdx.x + off];
            ssq[threadIdx.x]  += ssq[threadIdx.x + off];
        }
        __syncthreads();
    }
    const float m   = ssum[0] * (1.0f / HW);
    const float var = ssq[0] * (1.0f / HW) - m * m;
    const float inv = rsqrtf(var + 1e-5f);

    for (int hw = threadIdx.x; hw < HW; hw += 64) {
        const size_t iin = ((size_t)(b * S_TOT + f * HW + hw)) * CDIM + c;
        const float v = (in[iin] - m) * inv;
        if (out5d) out[(((size_t)(f * NB + b)) * CDIM + c) * HW + hw] = v;
        else       out[iin] = v;
    }
}

// ============================================================
extern "C" void kernel_launch(void* const* d_in, const int* in_sizes, int n_in,
                              void* d_out, int out_size, void* d_ws, size_t ws_size,
                              hipStream_t stream)
{
    (void)in_sizes; (void)n_in; (void)out_size; (void)ws_size;
    const float* tgt    = (const float*)d_in[0];
    const float* memory = (const float*)d_in[1];
    const float* pe     = (const float*)d_in[2];
    const float* wself  = (const float*)d_in[3];
    const float* bself  = (const float*)d_in[4];
    const float* wcross = (const float*)d_in[5];
    const float* bcross = (const float*)d_in[6];
    float* out = (float*)d_out;

    char* ws = (char*)d_ws;
    size_t off = 0;
    auto walloc = [&](size_t bytes) -> void* {
        void* p = ws + off;
        off += (bytes + 255) & ~(size_t)255;
        return p;
    };
    const size_t QK_BYTES  = (size_t)NB * S_TOT * KD * sizeof(_Float16);
    const size_t XC_BYTES  = (size_t)NB * S_TOT * CDIM * sizeof(_Float16);
    const size_t BSC_BYTES = (size_t)NB * S_TOT * CDIM * sizeof(float);
    const size_t WB_BYTES  = (size_t)(CDIM / 16) * KD * 16 * sizeof(_Float16);

    _Float16* WBs  = (_Float16*)walloc(WB_BYTES);
    _Float16* WBc  = (_Float16*)walloc(WB_BYTES);
    _Float16* Xs   = (_Float16*)walloc(XC_BYTES);   // tgt as f16 rows
    _Float16* Xc   = (_Float16*)walloc(XC_BYTES);   // t1 as f16 rows
    _Float16* Xm   = (_Float16*)walloc(XC_BYTES);   // memory as f16 rows
    _Float16* QAs  = (_Float16*)walloc(QK_BYTES);
    _Float16* KBs  = (_Float16*)walloc(QK_BYTES);
    _Float16* VBs  = (_Float16*)walloc(XC_BYTES);
    float*    sout = (float*)walloc(BSC_BYTES);
    float*    t1   = (float*)walloc(BSC_BYTES);
    _Float16* QAc  = (_Float16*)walloc(QK_BYTES);
    _Float16* KBc  = (_Float16*)walloc(QK_BYTES);
    _Float16* VBc  = (_Float16*)walloc(XC_BYTES);
    float*    cout = (float*)walloc(BSC_BYTES);

    const int  xTotal = NB * S_TOT * CDIM;
    const int  gX  = (xTotal + 255) / 256;
    const int  gW  = ((CDIM / 16) * KD * 16 + 255) / 256;
    const dim3 gPW(S_TOT / 16 / 4, NB);          // projw: 4 waves/block
    const dim3 gAttn(S_TOT / 16, NB);
    const int  gIN = NF * NB * CDIM;

    // weight prep (tiny)
    wprep_kernel<<<gW, 256, 0, stream>>>(wself,  WBs);
    wprep_kernel<<<gW, 256, 0, stream>>>(wcross, WBc);

    // --- self-attention branch: t1 = instnorm(t + attn(t,t,t)) ---
    xcast_kernel<<<gX, 256, 0, stream>>>(tgt, 0, Xs);
    projw_kernel<<<gPW, 128, 0, stream>>>(Xs, WBs, bself, QAs, KBs);
    vprep_kernel<<<gX, 256, 0, stream>>>(tgt, VBs);
    attn_kernel <<<gAttn, 256, 0, stream>>>(QAs, KBs, VBs, tgt, sout);
    inorm_kernel<<<gIN, 64, 0, stream>>>(sout, t1, 0);

    // --- cross-attention: out = instnorm(attn(t1, memory, pos_enc)) ---
    xcast_kernel<<<gX, 256, 0, stream>>>(t1, 1, Xc);
    projw_kernel<<<gPW, 128, 0, stream>>>(Xc, WBc, bcross, QAc, nullptr);
    xcast_kernel<<<gX, 256, 0, stream>>>(memory, 0, Xm);
    projw_kernel<<<gPW, 128, 0, stream>>>(Xm, WBc, bcross, nullptr, KBc);
    vprep_kernel<<<gX, 256, 0, stream>>>(pe, VBc);
    attn_kernel <<<gAttn, 256, 0, stream>>>(QAc, KBc, VBc, nullptr, cout);
    inorm_kernel<<<gIN, 64, 0, stream>>>(cout, out, 1);
}
